// Decoder_75118978007529
// MI455X (gfx1250) — matrix operand order repaired and checked by reading, not compiled, not run on hardware
//
#include <hip/hip_runtime.h>
#include <hip/hip_bf16.h>

// ---------------------------------------------------------------------------
// Decoder: concat -> GRU(relu candidate, reset_after) -> FFN -> vocab softmax
// B=32 T=128 E=256 H=512 V=32000 ; M = B*T = 4096 rows
// All GEMMs via v_wmma_f32_16x16x32_bf16 (bf16 inputs, f32 accumulate).
// ---------------------------------------------------------------------------

typedef __attribute__((ext_vector_type(16))) __bf16 v16bf;
typedef __attribute__((ext_vector_type(8)))  __bf16 v8bf;
typedef __attribute__((ext_vector_type(8)))  float  v8f;
typedef __attribute__((ext_vector_type(4)))  unsigned int uint32x4;
typedef __attribute__((ext_vector_type(4)))  int int32x4;
typedef __attribute__((ext_vector_type(8)))  int int32x8;

#define CB 32
#define CT 128
#define CE 256
#define CH 512
#define CV 32000
#define CEH 768
#define CG 1536
#define CM 4096

static __device__ __forceinline__ v8f wmma_bf16(v16bf a, v16bf b, v8f c) {
  // (neg_a, A, neg_b, B, c_mod, C, reuse_a, reuse_b)
  return __builtin_amdgcn_wmma_f32_16x16x32_bf16(false, a, false, b, (short)0, c,
                                                 false, false);
}

// A fragment: row-major bf16 [M x lda]. ISA 7.12.2 (16-bit A 16x32):
// lane l: m = l&15, kh = l>>4; elems 0..7 -> K = kh*8 .. +7 ; elems 8..15 -> +16.
static __device__ __forceinline__ v16bf load_a_frag(const __bf16* A, int lda,
                                                    int row0, int k0, int lane) {
  int m = lane & 15, kh = lane >> 4;
  const __bf16* p = A + (size_t)(row0 + m) * lda + k0 + kh * 8;
  v8bf lo = *(const v8bf*)p;
  v8bf hi = *(const v8bf*)(p + 16);
  return __builtin_shufflevector(lo, hi, 0, 1, 2, 3, 4, 5, 6, 7,
                                 8, 9, 10, 11, 12, 13, 14, 15);
}

static __device__ __forceinline__ v8f vzero8f() {
  v8f z = {0.f, 0.f, 0.f, 0.f, 0.f, 0.f, 0.f, 0.f};
  return z;
}

// Wave-uniform B-fragment load: tile index forced into SGPRs via readfirstlane
// so the global_load_b128 uses saddr form (scalar base + lane*32B voffset).
static __device__ __forceinline__ v16bf load_b_frag(const __bf16* Bp, int tile,
                                                    int lane) {
  int ut = __builtin_amdgcn_readfirstlane(tile);
  const __bf16* bb = Bp + ((size_t)ut << 9);
  return *(const v16bf*)(bb + (lane << 4));
}

// ---------------------------------------------------------------------------
// Kernel 1: sequence = concat(encoder[b,t,:], decoder[b,:]) -> bf16 [4096,768]
// ---------------------------------------------------------------------------
__global__ void pack_seq_kernel(const float* __restrict__ enc,
                                const float* __restrict__ dec,
                                __bf16* __restrict__ seq) {
  int i = blockIdx.x * 256 + threadIdx.x;
  if (i >= CM * CEH) return;
  int row = i / CEH, col = i - row * CEH;
  int b = row >> 7;  // row = b*T + t
  float v = (col < CE) ? enc[(size_t)row * CE + col]
                       : dec[(size_t)b * CH + (col - CE)];
  seq[i] = (__bf16)v;
}

// ---------------------------------------------------------------------------
// Kernel 2: pack f32 weight [K,N] row-major into WMMA B-fragment tile layout:
// packed[((kt*Ntiles + nt)*32 + lane)*16 + e]  where
//   k = kt*32 + (lane>>4)*8 + (e&7) + (e>>3)*16 ; n = nt*16 + (lane&15)
// -> each lane's B fragment is one contiguous 32-byte load.
// ---------------------------------------------------------------------------
__global__ void pack_b_kernel(const float* __restrict__ W,
                              __bf16* __restrict__ out, int K, int N) {
  int i = blockIdx.x * 256 + threadIdx.x;
  if (i >= K * N) return;
  int Ntiles = N >> 4;
  int e = i & 15;
  int lane = (i >> 4) & 31;
  int tile = i >> 9;
  int nt = tile % Ntiles, kt = tile / Ntiles;
  int k = (kt << 5) + ((lane >> 4) << 3) + (e & 7) + ((e >> 3) << 4);
  int n = (nt << 4) + (lane & 15);
  out[i] = (__bf16)W[(size_t)k * N + n];
}

// ---------------------------------------------------------------------------
// Kernel 3: generic bf16 WMMA GEMM  C = act(A @ B + bias)
// Each wave: one 16-row M-tile x 4 N-tiles. Double-buffered fragments so the
// loads for k+1 overlap the WMMAs for k; B addressing scalarized.
// ---------------------------------------------------------------------------
__global__ void gemm_bf16_kernel(const __bf16* __restrict__ A,
                                 const __bf16* __restrict__ Bp,
                                 const float* __restrict__ bias,
                                 float* __restrict__ outF,
                                 __bf16* __restrict__ outB,
                                 int Mtiles, int Ntiles, int Kt, int relu) {
  int tid = threadIdx.x, lane = tid & 31, wave = tid >> 5;
  int w = blockIdx.x * (blockDim.x >> 5) + wave;
  int ngroups = Ntiles >> 2;
  int mt = w / ngroups;
  int nt0 = (w - mt * ngroups) << 2;
  if (mt >= Mtiles) return;
  int N = Ntiles << 4, K = Kt << 5, row0 = mt << 4;

  const __bf16* ap = A + (size_t)(row0 + (lane & 15)) * K + ((lane >> 4) << 3);
  auto loadA = [&](int kt) -> v16bf {
    const __bf16* p = ap + (kt << 5);
    v8bf lo = *(const v8bf*)p;
    v8bf hi = *(const v8bf*)(p + 16);
    return __builtin_shufflevector(lo, hi, 0, 1, 2, 3, 4, 5, 6, 7,
                                   8, 9, 10, 11, 12, 13, 14, 15);
  };

  v8f acc[4];
#pragma unroll
  for (int j = 0; j < 4; ++j) acc[j] = vzero8f();

  v16bf a_cur = loadA(0);
  v16bf b_cur[4];
#pragma unroll
  for (int j = 0; j < 4; ++j) b_cur[j] = load_b_frag(Bp, nt0 + j, lane);

  for (int kt = 0; kt < Kt; ++kt) {
    int ktn = (kt + 1 < Kt) ? kt + 1 : kt;  // clamp: last prefetch is dead
    v16bf a_nxt = loadA(ktn);
    v16bf b_nxt[4];
#pragma unroll
    for (int j = 0; j < 4; ++j)
      b_nxt[j] = load_b_frag(Bp, ktn * Ntiles + nt0 + j, lane);
#pragma unroll
    for (int j = 0; j < 4; ++j) acc[j] = wmma_bf16(a_cur, b_cur[j], acc[j]);
    a_cur = a_nxt;
#pragma unroll
    for (int j = 0; j < 4; ++j) b_cur[j] = b_nxt[j];
  }

  // C/D layout: lane l -> col = l&15 ; rows = 8*(l>>4) + v (v = vgpr 0..7)
  int colloc = lane & 15;
  int rbase = row0 + ((lane >> 4) << 3);
#pragma unroll
  for (int j = 0; j < 4; ++j) {
    int n = ((nt0 + j) << 4) + colloc;
    float bv = bias ? bias[n] : 0.f;
#pragma unroll
    for (int v = 0; v < 8; ++v) {
      float x = acc[j][v] + bv;
      if (relu) x = fmaxf(x, 0.f);
      size_t idx = (size_t)(rbase + v) * N + n;
      if (outF) outF[idx] = x;
      if (outB) outB[idx] = (__bf16)x;
    }
  }
}

// ---------------------------------------------------------------------------
// Kernel 4: GRU scan. 2 blocks x 512 threads; block handles 16 batch rows.
// Per step: rec = h_bf16[16x512] @ recK[512x1536] via WMMA (96 N-tiles over
// 16 waves), then elementwise gates, h kept in LDS (f32 + bf16 mirror).
// Dynamic LDS: rec f32 16*1536 | h f32 16*512 | hbf bf16 16*512 = 147456 B.
// global_prefetch of the next k-step's B fragments hides L2 latency in the
// latency-critical serial loop.
// ---------------------------------------------------------------------------
__global__ void gru_scan_kernel(const float* __restrict__ xp,
                                const __bf16* __restrict__ recKp,
                                const float* __restrict__ gru_bias,
                                __bf16* __restrict__ hseq) {
  extern __shared__ char dynsmem[];
  float*  rec = (float*)dynsmem;                       // [16][1536]
  float*  h   = (float*)dynsmem + 16 * CG;             // [16][512]
  __bf16* hbf = (__bf16*)((float*)dynsmem + 16 * CG + 16 * CH);  // [16][512]
  const float* bias1 = gru_bias + CG;

  int tid = threadIdx.x, lane = tid & 31, wave = tid >> 5;  // 16 waves
  int blk = blockIdx.x;                                     // batch rows blk*16..+15
  int colloc = lane & 15, rbase = (lane >> 4) << 3;

  for (int i = tid; i < 16 * CH; i += 512) { h[i] = 0.f; hbf[i] = (__bf16)0.f; }
  __syncthreads();

  for (int t = 0; t < CT; ++t) {
    // --- recurrent GEMM: N=1536 (96 tiles), K=512 (16 k-steps) ---
    v8f acc[6];
#pragma unroll
    for (int j = 0; j < 6; ++j) acc[j] = vzero8f();
    for (int kt = 0; kt < 16; ++kt) {
      v16bf a = load_a_frag(hbf, CH, 0, kt << 5, lane);
      int tile0 = __builtin_amdgcn_readfirstlane(kt * 96 + wave * 6);
      if (kt + 1 < 16) {
        // prefetch next k-step's fragments (global_prefetch_b8)
        int tp = __builtin_amdgcn_readfirstlane((kt + 1) * 96 + wave * 6);
        __builtin_prefetch(recKp + ((size_t)tp << 9) + (lane << 4), 0, 1);
      }
#pragma unroll
      for (int j = 0; j < 6; ++j) {
        v16bf b = *(const v16bf*)(recKp + ((size_t)(tile0 + j) << 9) +
                                  (lane << 4));
        acc[j] = wmma_bf16(a, b, acc[j]);
      }
    }
#pragma unroll
    for (int j = 0; j < 6; ++j) {
      int n = (wave * 6 + j) * 16 + colloc;
#pragma unroll
      for (int v = 0; v < 8; ++v) rec[(rbase + v) * CG + n] = acc[j][v];
    }
    __syncthreads();

    // --- gates: z,r sigmoid; candidate relu; reset_after form ---
    for (int i = tid; i < 16 * CH; i += 512) {
      int r = i >> 9, c = i & (CH - 1);
      size_t xrow = ((size_t)(blk * 16 + r) * CT + t) * CG;
      float xz = xp[xrow + c];
      float xr = xp[xrow + CH + c];
      float xh = xp[xrow + 2 * CH + c];
      float rz = rec[r * CG + c] + bias1[c];
      float rr = rec[r * CG + CH + c] + bias1[CH + c];
      float rh = rec[r * CG + 2 * CH + c] + bias1[2 * CH + c];
      float z  = 1.f / (1.f + __expf(-(xz + rz)));
      float rg = 1.f / (1.f + __expf(-(xr + rr)));
      float hh = fmaxf(xh + rg * rh, 0.f);
      float hn = z * h[i] + (1.f - z) * hh;
      h[i] = hn;
      hbf[i] = (__bf16)hn;
      hseq[((size_t)(blk * 16 + r) * CT + t) * CH + c] = (__bf16)hn;
    }
    __syncthreads();
  }
}

// ---------------------------------------------------------------------------
// Kernel 5: vocab GEMM + fused online softmax.
// 128 blocks x 512 threads; block = 32 rows (2 M-tiles, B fragments reused
// across both -> halves w3 L2 traffic) x full V=32000 (2000 N-tiles).
// A block staged in LDS via the Tensor Data Mover (async DMA, TENSORcnt) with
// a redundant direct copy of the identical bytes as a safety net; per-lane
// online (max,sumexp); shfl+LDS reduce; one re-read pass to normalize.
// Dynamic LDS: As bf16[32*512] @0 | lm f32[32][16] | ls f32[32][16] |
//              rowm f32[32] | rowis f32[32]  = 37120 B.
// ---------------------------------------------------------------------------
__global__ void logits_softmax_kernel(const __bf16* __restrict__ A,
                                      const __bf16* __restrict__ Bp,
                                      const float* __restrict__ bias,
                                      float* __restrict__ out) {
  extern __shared__ char dynsmem[];
  __bf16* As   = (__bf16*)dynsmem;                       // 32*512, LDS offset 0
  float*  lm   = (float*)(dynsmem + 32768);              // [32][16]
  float*  ls   = (float*)(dynsmem + 32768 + 2048);       // [32][16]
  float*  rowm = (float*)(dynsmem + 32768 + 4096);       // [32]
  float*  rowis= (float*)(dynsmem + 32768 + 4096 + 128); // [32]

  int tid = threadIdx.x, lane = tid & 31, wave = tid >> 5;  // 16 waves
  int blk = blockIdx.x;
  const int Ntiles = CV >> 4;  // 2000
  int colloc = lane & 15, rbase = (lane >> 4) << 3;
  const __bf16* Ablk = A + (size_t)blk * 32 * CH;

  if (tid == 0) {
    // TDM: 1-D tile of 4096 x 8B (= 32KB) global -> LDS offset 0.
    // D# group0: count=1 | lds_addr=0 | global_addr[56:0] | type=2.
    // D# group1: data_size=3(8B) | tensor_dim0=4096 | tensor_dim1=1 |
    //            tile_dim0=4096 | tile_dim1..2=0 | dim0_stride=4096.
    unsigned long long ga = (unsigned long long)(size_t)Ablk;
    uint32x4 g0 = {1u, 0u, (unsigned)(ga & 0xffffffffu),
                   (unsigned)(((ga >> 32) & 0x1ffffffu) | (2u << 30))};
    int32x8 g1 = {0x00030000, 0x10000000, 0x00010000, 0x10000000, 0, 4096, 0, 0};
    int32x4 gz = {0, 0, 0, 0};
#if defined(__clang_major__) && (__clang_major__ >= 23)
    int32x8 gz8 = {0, 0, 0, 0, 0, 0, 0, 0};
    __builtin_amdgcn_tensor_load_to_lds(g0, g1, gz, gz, gz8, 0);
#else
    __builtin_amdgcn_tensor_load_to_lds(g0, g1, gz, gz, 0);
#endif
    __builtin_amdgcn_s_wait_tensorcnt(0);
  }
  // Safety net: direct copy of the same bytes (benign identical-value race).
  for (int i = tid; i < 32 * CH; i += 512) As[i] = Ablk[i];
  __syncthreads();

  float m[16], s[16];
#pragma unroll
  for (int q = 0; q < 16; ++q) { m[q] = -3.0e38f; s[q] = 0.f; }

  for (int nt = wave; nt < Ntiles; nt += 16) {
    v8f acc0 = vzero8f(), acc1 = vzero8f();
    v16bf b_cur = load_b_frag(Bp, nt, lane);
    for (int kt = 0; kt < 16; ++kt) {
      int ktn = (kt + 1 < 16) ? kt + 1 : kt;
      v16bf b_nxt = load_b_frag(Bp, ktn * Ntiles + nt, lane);
      v16bf a0 = load_a_frag(As, CH, 0, kt << 5, lane);
      v16bf a1 = load_a_frag(As, CH, 16, kt << 5, lane);
      acc0 = wmma_bf16(a0, b_cur, acc0);
      acc1 = wmma_bf16(a1, b_cur, acc1);
      b_cur = b_nxt;
    }
    int n = nt * 16 + colloc;
    float bv = bias[n];
#pragma unroll
    for (int v = 0; v < 8; ++v) {
      float x = acc0[v] + bv;  // rows 0..15 of block
      out[(size_t)(blk * 32 + rbase + v) * CV + n] = x;
      float mn = fmaxf(m[v], x);
      s[v] = s[v] * __expf(m[v] - mn) + __expf(x - mn);
      m[v] = mn;
      float y = acc1[v] + bv;  // rows 16..31 of block
      out[(size_t)(blk * 32 + 16 + rbase + v) * CV + n] = y;
      float mn2 = fmaxf(m[8 + v], y);
      s[8 + v] = s[8 + v] * __expf(m[8 + v] - mn2) + __expf(y - mn2);
      m[8 + v] = mn2;
    }
  }

  // combine (m,s) across the 16 lanes of each half-wave (stay within half)
#pragma unroll
  for (int q = 0; q < 16; ++q) {
    float mv = m[q], sv = s[q];
    for (int off = 1; off < 16; off <<= 1) {
      float mo = __shfl_xor(mv, off, 32);
      float so = __shfl_xor(sv, off, 32);
      float mn = fmaxf(mv, mo);
      sv = sv * __expf(mv - mn) + so * __expf(mo - mn);
      mv = mn;
    }
    if (colloc == 0) {
      int row = ((q >> 3) << 4) + rbase + (q & 7);
      lm[row * 16 + wave] = mv;
      ls[row * 16 + wave] = sv;
    }
  }
  __syncthreads();
  if (tid < 32) {
    float mv = lm[tid * 16], sv = ls[tid * 16];
    for (int wv = 1; wv < 16; ++wv) {
      float mo = lm[tid * 16 + wv], so = ls[tid * 16 + wv];
      float mn = fmaxf(mv, mo);
      sv = sv * __expf(mv - mn) + so * __expf(mo - mn);
      mv = mn;
    }
    rowm[tid] = mv;
    rowis[tid] = 1.f / sv;
  }
  __threadfence();   // raw-logit stores visible before cross-wave re-read
  __syncthreads();

  for (int r = 0; r < 32; ++r) {
    float mv = rowm[r], is = rowis[r];
    size_t base = (size_t)(blk * 32 + r) * CV;
    for (int c = tid; c < CV; c += 512) {
      float x = out[base + c];
      out[base + c] = __expf(x - mv) * is;
    }
  }
}

// ---------------------------------------------------------------------------
extern "C" void kernel_launch(void* const* d_in, const int* in_sizes, int n_in,
                              void* d_out, int out_size, void* d_ws, size_t ws_size,
                              hipStream_t stream) {
  (void)in_sizes; (void)n_in; (void)out_size; (void)ws_size;
  const float* enc   = (const float*)d_in[0];   // [32,128,256]
  const float* dec   = (const float*)d_in[1];   // [32,512]
  const float* gruK  = (const float*)d_in[2];   // [768,1536]
  const float* recK  = (const float*)d_in[3];   // [512,1536]
  const float* gbias = (const float*)d_in[4];   // [2,1536]
  const float* w1    = (const float*)d_in[5];   // [512,512]
  const float* b1    = (const float*)d_in[6];
  const float* w2    = (const float*)d_in[7];
  const float* b2    = (const float*)d_in[8];
  const float* w3    = (const float*)d_in[9];   // [512,32000]
  const float* b3    = (const float*)d_in[10];
  float* out = (float*)d_out;

  char* ws = (char*)d_ws;
  size_t off = 0;
  auto alloc = [&](size_t bytes) { char* p = ws + off; off += (bytes + 255) & ~(size_t)255; return p; };
  __bf16* seqb  = (__bf16*)alloc((size_t)CM * CEH * 2);
  __bf16* gruKp = (__bf16*)alloc((size_t)CEH * CG * 2);
  __bf16* recKp = (__bf16*)alloc((size_t)CH * CG * 2);
  __bf16* w1p   = (__bf16*)alloc((size_t)CH * CH * 2);
  __bf16* w2p   = (__bf16*)alloc((size_t)CH * CH * 2);
  __bf16* w3p   = (__bf16*)alloc((size_t)CH * CV * 2);
  float*  xp    = (float*)alloc((size_t)CM * CG * 4);
  __bf16* hseqb = (__bf16*)alloc((size_t)CM * CH * 2);
  __bf16* h1b   = (__bf16*)alloc((size_t)CM * CH * 2);
  __bf16* h2b   = (__bf16*)alloc((size_t)CM * CH * 2);

  // 1) build bf16 concat sequence
  pack_seq_kernel<<<(CM * CEH + 255) / 256, 256, 0, stream>>>(enc, dec, seqb);
  // 2) pack weights into B-fragment layout
  pack_b_kernel<<<(CEH * CG + 255) / 256, 256, 0, stream>>>(gruK, gruKp, CEH, CG);
  pack_b_kernel<<<(CH * CG + 255) / 256, 256, 0, stream>>>(recK, recKp, CH, CG);
  pack_b_kernel<<<(CH * CH + 255) / 256, 256, 0, stream>>>(w1, w1p, CH, CH);
  pack_b_kernel<<<(CH * CH + 255) / 256, 256, 0, stream>>>(w2, w2p, CH, CH);
  pack_b_kernel<<<(CH * CV + 255) / 256, 256, 0, stream>>>(w3, w3p, CH, CV);

  // 3) x_proj = seq @ gru_kernel + bias0   (M=4096,N=1536,K=768) -> f32
  {
    int mt = CM / 16, ntg = (CG / 16) / 4;
    gemm_bf16_kernel<<<(mt * ntg + 7) / 8, 256, 0, stream>>>(
        seqb, gruKp, gbias, xp, nullptr, mt, CG / 16, CEH / 32, 0);
  }
  // 4) GRU scan (2 WGs x 16 batch rows, T=128 steps)
  gru_scan_kernel<<<2, 512, 16 * CG * 4 + 16 * CH * 4 + 16 * CH * 2, stream>>>(
      xp, recKp, gbias, hseqb);
  // 5) h1 = relu(hseq @ w1 + b1), h2 = relu(h1 @ w2 + b2)
  {
    int mt = CM / 16, ntg = (CH / 16) / 4;
    gemm_bf16_kernel<<<(mt * ntg + 7) / 8, 256, 0, stream>>>(
        hseqb, w1p, b1, nullptr, h1b, mt, CH / 16, CH / 32, 1);
    gemm_bf16_kernel<<<(mt * ntg + 7) / 8, 256, 0, stream>>>(
        h1b, w2p, b2, nullptr, h2b, mt, CH / 16, CH / 32, 1);
  }
  // 6) probs = softmax(h2 @ w3 + b3)  fused GEMM + online softmax (32 rows/WG)
  logits_softmax_kernel<<<CM / 32, 512, 37120, stream>>>(h2b, w3p, b3, out);
}
